// class_MMD_loss_65541200937314
// MI455X (gfx1250) — matrix operand order repaired and checked by reading, not compiled, not run on hardware
//
#include <hip/hip_runtime.h>
#include <hip/hip_bf16.h>

// ---------------------------------------------------------------------------
// MMD loss for MI455X (gfx1250).
// GEMM total@total^T (N=4096, D=2048) via v_wmma_f32_16x16x32_bf16, upper
// block-triangle only (D2 symmetric, every reduction over it is symmetric ->
// off-diagonal blocks weighted x2). Tile data is streamed global->LDS by the
// Tensor Data Mover (tensor_load_to_lds + TENSORcnt) with double-buffered LDS,
// one workgroup barrier per K-step. Per-class sums use LDS float atomics.
// Workspace: D2 upper tiles (67MB region), Tbf (16.8MB), small accumulators.
// ---------------------------------------------------------------------------

#define NS   2048
#define NN   4096
#define DD   2048
#define CLS  31
#define NB   32          // 4096 / 128 tiles per dim
#define NPAIR 528        // NB*(NB+1)/2 upper-triangle tiles

typedef __attribute__((ext_vector_type(16))) __bf16     v16bf;
typedef __attribute__((ext_vector_type(8)))  float      v8f;
typedef __attribute__((ext_vector_type(4)))  unsigned   u32x4;
typedef __attribute__((ext_vector_type(8)))  int        i32x8;
typedef __attribute__((ext_vector_type(4)))  int        i32x4;
typedef __attribute__((ext_vector_type(4)))  float      f32x4;

struct P2 { u32x4 lo, hi; };   // 32 bytes -> bit_cast to v16bf

__device__ __forceinline__ unsigned short f32_to_bf16(float f) {
  unsigned u = __float_as_uint(f);
  unsigned r = u + 0x7FFFu + ((u >> 16) & 1u);   // round-to-nearest-even
  return (unsigned short)(r >> 16);
}

// ---- TDM: load a 128x32 bf16 tile (row stride DD elems) into LDS ----------
// D# per CDNA5 ISA ch.8: group0 {count, lds_addr, global_addr, type=2},
// group1 {data_size=2B, tensor dims, tile dims, dim0 stride}; groups 2/3 = 0
// (2-D tile). Issued by one wave; tracked on TENSORcnt.
__device__ __forceinline__ void tdm_load_tile(const unsigned short* gtile,
                                              unsigned lds_bytes) {
  unsigned long long ga = (unsigned long long)(uintptr_t)gtile;
  u32x4 g0;
  g0[0] = 1u;                                            // count=1
  g0[1] = lds_bytes;                                     // lds_addr
  g0[2] = (unsigned)(ga & 0xFFFFFFFFu);                  // global_addr[31:0]
  g0[3] = (unsigned)((ga >> 32) & 0x01FFFFFFu) | (2u << 30); // addr[56:32]|type=2
  i32x8 g1;
  g1[0] = 0x00010000;                                    // data_size=1 (2 bytes)
  g1[1] = (int)((DD & 0xFFFFu) << 16);                   // tensor_dim0 lo16
  g1[2] = (int)(((DD >> 16) & 0xFFFFu) | ((NN & 0xFFFFu) << 16)); // d0 hi|d1 lo
  g1[3] = (int)(((NN >> 16) & 0xFFFFu) | (32u << 16));   // d1 hi | tile_dim0=32
  g1[4] = 128;                                           // tile_dim1=128
  g1[5] = DD;                                            // dim0_stride lo32
  g1[6] = 0;
  g1[7] = 0;
  i32x4 z4 = {0, 0, 0, 0};
#if defined(__clang_major__) && (__clang_major__ >= 23)
  i32x8 z8 = {0, 0, 0, 0, 0, 0, 0, 0};
  __builtin_amdgcn_tensor_load_to_lds(g0, g1, z4, z4, z8, 0);
#else
  __builtin_amdgcn_tensor_load_to_lds(g0, g1, z4, z4, 0);
#endif
}

// ---- K0: zero the small accumulators (harness poisons ws with 0xAA) -------
__global__ void k_zero(int* nsrc, int* ntgt, float* l2sum, float* contrib) {
  int t = threadIdx.x;
  if (t < CLS) { nsrc[t] = 0; ntgt[t] = 0; l2sum[t] = 0.f; contrib[t] = 0.f; }
}

// ---- K1: labels = [s_label, argmax(t_label)] + per-class domain counts ----
__global__ void k_labels(const int* __restrict__ s_label,
                         const float* __restrict__ t_label,
                         int* __restrict__ labels,
                         int* __restrict__ nsrc, int* __restrict__ ntgt) {
  int i = blockIdx.x * blockDim.x + threadIdx.x;
  if (i < NS) {
    int c = s_label[i];
    labels[i] = c;
    atomicAdd(&nsrc[c], 1);
  } else if (i < NN) {
    const float* row = t_label + (size_t)(i - NS) * CLS;
    float best = row[0]; int bi = 0;
    for (int c = 1; c < CLS; ++c) { float v = row[c]; if (v > best) { best = v; bi = c; } }
    labels[i] = bi;
    atomicAdd(&ntgt[bi], 1);
  }
}

// ---- K2: bf16 convert + exact f32 row squared-norms -----------------------
__global__ __launch_bounds__(256) void k_convert(const float* __restrict__ src,
                                                 const float* __restrict__ tgt,
                                                 unsigned short* __restrict__ Tbf,
                                                 float* __restrict__ sq) {
  __shared__ float red[256];
  int row = blockIdx.x;
  const float* rp = (row < NS) ? (src + (size_t)row * DD)
                               : (tgt + (size_t)(row - NS) * DD);
  float acc = 0.f;
  for (int c = threadIdx.x; c < DD; c += 256) {
    float v = rp[c];
    acc += v * v;
    Tbf[(size_t)row * DD + c] = f32_to_bf16(v);
  }
  red[threadIdx.x] = acc;
  __syncthreads();
  for (int s = 128; s > 0; s >>= 1) {
    if (threadIdx.x < s) red[threadIdx.x] += red[threadIdx.x + s];
    __syncthreads();
  }
  if (threadIdx.x == 0) sq[row] = red[0];
}

// ---- K3: WMMA GEMM (upper block-triangle), TDM-fed, double-buffered -------
// 256 threads = 8 wave32. Wave w: wm=w&1 (64 rows), wn=w>>1 (32 cols);
// per wave 4x2 subtiles of 16x16; K stepped by 32.
__global__ __launch_bounds__(256, 1) void k_gemm(
    const unsigned short* __restrict__ Tbf,
    const float* __restrict__ sq, const int* __restrict__ labels,
    float* __restrict__ D2, float* __restrict__ l2sum) {
  __shared__ unsigned short lA[2][128 * 32];   // 2 x 8KB
  __shared__ unsigned short lB[2][128 * 32];   // 2 x 8KB
  __shared__ float binL[CLS];

  // linear block -> (bm, bn) with bm <= bn
  int rem = blockIdx.x, bm = 0, rowlen = NB;
  while (rem >= rowlen) { rem -= rowlen; rowlen--; bm++; }
  int bn = bm + rem;

  if (threadIdx.x < CLS) binL[threadIdx.x] = 0.f;

  int t    = threadIdx.x;
  int lane = t & 31;
  int w    = t >> 5;
  int wm   = w & 1;        // 0..1
  int wn   = w >> 1;       // 0..3
  int lr   = lane & 15;
  int hi   = lane >> 4;    // K-half select: k = 16*hi + s (contiguous order)

  const unsigned short* gA = Tbf + (size_t)(bm * 128) * DD;
  const unsigned short* gB = Tbf + (size_t)(bn * 128) * DD;
  bool issuer = (w == 0);  // wave-uniform: one wave drives the TDM

  v8f acc[4][2];
  for (int sm = 0; sm < 4; ++sm)
    for (int sn = 0; sn < 2; ++sn)
      for (int r = 0; r < 8; ++r) acc[sm][sn][r] = 0.f;

  // prologue: stage K-slab 0 into buffer 0
  if (issuer) {
    tdm_load_tile(gA, (unsigned)(uintptr_t)&lA[0][0]);
    tdm_load_tile(gB, (unsigned)(uintptr_t)&lB[0][0]);
    __builtin_amdgcn_s_wait_tensorcnt(0);
  }
  __syncthreads();

  for (int kk = 0; kk < DD; kk += 32) {
    int b = (kk >> 5) & 1;
    // stage next K-slab into the other buffer while we compute
    if (issuer && (kk + 32) < DD) {
      tdm_load_tile(gA + kk + 32, (unsigned)(uintptr_t)&lA[1 - b][0]);
      tdm_load_tile(gB + kk + 32, (unsigned)(uintptr_t)&lB[1 - b][0]);
    }

    // gather all fragments first, then 8 back-to-back WMMAs
    const unsigned short* As = &lA[b][0];
    const unsigned short* Bs = &lB[b][0];
    v16bf fb[2], fa[4];
    for (int sn = 0; sn < 2; ++sn) {
      // B 32x16: lane = col N=lr, K = hi*16 .. +15 (32B contiguous)
      const unsigned short* pb = Bs + (wn * 32 + sn * 16 + lr) * 32 + hi * 16;
      P2 p; p.lo = *(const u32x4*)pb; p.hi = *(const u32x4*)(pb + 8);
      fb[sn] = __builtin_bit_cast(v16bf, p);
    }
    for (int sm = 0; sm < 4; ++sm) {
      // A 16x32: lane = row M=lr, K = hi*16 .. +15 (32B contiguous)
      const unsigned short* pa = As + (wm * 64 + sm * 16 + lr) * 32 + hi * 16;
      P2 p; p.lo = *(const u32x4*)pa; p.hi = *(const u32x4*)(pa + 8);
      fa[sm] = __builtin_bit_cast(v16bf, p);
    }
    for (int sm = 0; sm < 4; ++sm)
      for (int sn = 0; sn < 2; ++sn)
        acc[sm][sn] = __builtin_amdgcn_wmma_f32_16x16x32_bf16(
            false, fa[sm], false, fb[sn], (short)0, acc[sm][sn], false, false);

#if __has_builtin(__builtin_amdgcn_sched_group_barrier)
    // Pin schedule: all 12 DS reads (mask 0x100) issue before the 8 WMMAs
    // (mask 0x8) -> one ds clause + single wait, no per-pair LDS bubbles.
    __builtin_amdgcn_sched_group_barrier(0x100, 12, 0);
    __builtin_amdgcn_sched_group_barrier(0x008, 8, 0);
#endif

    if (issuer) __builtin_amdgcn_s_wait_tensorcnt(0);  // next slab landed
    __syncthreads();                                    // everyone done w/ buf b
  }

  // Epilogue: D2 = sq_i + sq_j - 2*G; store; fused per-class l2 accumulation.
  float wgt = (bm < bn) ? 2.f : 1.f;   // symmetry weight
  for (int sm = 0; sm < 4; ++sm) {
    int ibase = bm * 128 + wm * 64 + sm * 16 + hi * 8;   // D row = r + hi*8
    for (int sn = 0; sn < 2; ++sn) {
      int j   = bn * 128 + wn * 32 + sn * 16 + lr;       // D col = lane&15
      float sqj = sq[j];
      int   lj  = labels[j];
      for (int r = 0; r < 8; ++r) {
        int i = ibase + r;
        float d2 = sq[i] + sqj - 2.0f * acc[sm][sn][r];
        D2[(size_t)i * NN + j] = d2;
        if (labels[i] == lj) atomicAdd(&binL[lj], wgt * d2);  // ds_add_f32
      }
    }
  }
  __syncthreads();
  if (threadIdx.x < CLS) atomicAdd(&l2sum[threadIdx.x], binL[threadIdx.x]);
}

// ---- K4: per-class bandwidth -> 5 reciprocal bandwidths -------------------
__global__ void k_bw(const float* __restrict__ l2sum,
                     const int* __restrict__ nsrc, const int* __restrict__ ntgt,
                     float* __restrict__ invbw5, float* __restrict__ inv_nn,
                     int* __restrict__ presentf) {
  int c = threadIdx.x;
  if (c >= CLS) return;
  float ns = (float)nsrc[c], nt = (float)ntgt[c];
  int present = (ns > 0.f) && (nt > 0.f);
  float n = ns + nt;
  float denom = fmaxf(n * n - n, 1.f);
  float bw = (l2sum[c] / denom) * 0.25f;          // / bw_scale (2^(5//2) = 4)
  bw = present ? fmaxf(bw, 1e-20f) : 1.0f;
  float ib = 1.0f / bw;
  for (int k = 0; k < 5; ++k) { invbw5[c * 5 + k] = ib; ib *= 0.5f; }  // KERNEL_MUL=2
  inv_nn[c]   = 1.0f / fmaxf(n * n, 1.f);
  presentf[c] = present;
}

// ---- K5: exp pass over stored upper-triangle D2 tiles ---------------------
__global__ __launch_bounds__(256) void k_kernel_sum(
    const float* __restrict__ D2, const int* __restrict__ labels,
    const float* __restrict__ invbw5, float* __restrict__ contrib) {
  __shared__ float binC[CLS];
  if (threadIdx.x < CLS) binC[threadIdx.x] = 0.f;
  __syncthreads();

  int rem = blockIdx.x, bm = 0, rowlen = NB;
  while (rem >= rowlen) { rem -= rowlen; rowlen--; bm++; }
  int bn = bm + rem;
  float wgt = (bm < bn) ? 2.f : 1.f;

  for (int e = threadIdx.x; e < 128 * 32; e += 256) {   // 128x128 elems as float4
    int r  = e >> 5;                 // row in tile
    int c4 = (e & 31) << 2;          // col (x4)
    int i  = bm * 128 + r;
    int li = labels[i];
    int j0 = bn * 128 + c4;
    f32x4 d4 = *(const f32x4*)(D2 + (size_t)i * NN + j0);
    i32x4 l4 = *(const i32x4*)(labels + j0);
    float si = (i < NS) ? 1.f : -1.f;
    for (int q = 0; q < 4; ++q) {
      if (l4[q] == li) {
        int j = j0 + q;
        float d2 = d4[q];
        const float* ib = invbw5 + li * 5;
        float ks = __expf(-d2 * ib[0]) + __expf(-d2 * ib[1]) + __expf(-d2 * ib[2])
                 + __expf(-d2 * ib[3]) + __expf(-d2 * ib[4]);
        float sj = (j < NS) ? 1.f : -1.f;
        atomicAdd(&binC[li], wgt * si * sj * ks);
      }
    }
  }
  __syncthreads();
  if (threadIdx.x < CLS) atomicAdd(&contrib[threadIdx.x], binC[threadIdx.x]);
}

// ---- K6: final scalar -----------------------------------------------------
__global__ void k_final(const float* __restrict__ contrib,
                        const float* __restrict__ inv_nn,
                        const int* __restrict__ presentf, float* __restrict__ out) {
  if (threadIdx.x == 0) {
    float loss = 0.f, cnt = 0.f;
    for (int c = 0; c < CLS; ++c)
      if (presentf[c]) { loss += contrib[c] * inv_nn[c]; cnt += 1.f; }
    out[0] = loss / cnt;
  }
}

// ---------------------------------------------------------------------------
extern "C" void kernel_launch(void* const* d_in, const int* in_sizes, int n_in,
                              void* d_out, int out_size, void* d_ws, size_t ws_size,
                              hipStream_t stream) {
  const float* source  = (const float*)d_in[0];
  const float* target  = (const float*)d_in[1];
  const int*   s_label = (const int*)d_in[2];
  const float* t_label = (const float*)d_in[3];
  float* out = (float*)d_out;

  // workspace carve-out (256B aligned slabs)
  size_t off = 0;
  char* base = (char*)d_ws;
  auto carve = [&](size_t bytes) -> void* {
    void* p = base + off;
    off += (bytes + 255) & ~(size_t)255;
    return p;
  };
  float*          D2      = (float*)carve((size_t)NN * NN * sizeof(float));
  unsigned short* Tbf     = (unsigned short*)carve((size_t)NN * DD * sizeof(unsigned short));
  float*          sq      = (float*)carve(NN * sizeof(float));
  int*            labels  = (int*)carve(NN * sizeof(int));
  int*            nsrc    = (int*)carve(CLS * sizeof(int));
  int*            ntgt    = (int*)carve(CLS * sizeof(int));
  float*          l2sum   = (float*)carve(CLS * sizeof(float));
  float*          contrib = (float*)carve(CLS * sizeof(float));
  float*          invbw5  = (float*)carve(CLS * 5 * sizeof(float));
  float*          inv_nn  = (float*)carve(CLS * sizeof(float));
  int*            present = (int*)carve(CLS * sizeof(int));

  k_zero<<<1, 32, 0, stream>>>(nsrc, ntgt, l2sum, contrib);
  k_labels<<<(NN + 255) / 256, 256, 0, stream>>>(s_label, t_label, labels, nsrc, ntgt);
  k_convert<<<NN, 256, 0, stream>>>(source, target, Tbf, sq);
  k_gemm<<<NPAIR, 256, 0, stream>>>(Tbf, sq, labels, D2, l2sum);
  k_bw<<<1, 32, 0, stream>>>(l2sum, nsrc, ntgt, invbw5, inv_nn, present);
  k_kernel_sum<<<NPAIR, 256, 0, stream>>>(D2, labels, invbw5, contrib);
  k_final<<<1, 32, 0, stream>>>(contrib, inv_nn, present, out);
}
